// PerceiverAttention_22660247453915
// MI455X (gfx1250) — compile-verified, matrix operand order checked by reading, not run on hardware
//
#include <hip/hip_runtime.h>
#include <hip/hip_bf16.h>
#include <math.h>

typedef __attribute__((ext_vector_type(16))) __bf16 v16bf;
typedef __attribute__((ext_vector_type(8)))  float  v8f;

constexpr int kB  = 8;
constexpr int kN  = 4096;
constexpr int kM  = 256;
constexpr int kD  = 1024;
constexpr int kH  = 16;
constexpr int kHD = 64;
constexpr int kInner = kH * kHD;   // 1024
constexpr int kNM = kN + kM;       // 4352

// ---------------- CDNA5 async global->LDS copy (ASYNCcnt path) ----------------

__device__ __forceinline__ void async_load_b128(const void* gptr, unsigned lds_bytes) {
  // GLOBAL_LOAD_ASYNC_TO_LDS_B128: vdst = LDS byte address, vaddr = 64-bit VA.
  asm volatile("global_load_async_to_lds_b128 %0, %1, off"
               :: "v"(lds_bytes), "v"((unsigned long long)(size_t)gptr)
               : "memory");
}
__device__ __forceinline__ void wait_async0() {
  asm volatile("s_wait_asynccnt 0x0" ::: "memory");
}
__device__ __forceinline__ unsigned lds_off(const void* p) {
  return (unsigned)(size_t)p;  // flat LDS aperture: low 32 bits = LDS offset
}

// ---------------- WMMA helpers (CDNA5 wave32 layouts) ----------------

__device__ __forceinline__ int kpack_base(int s, int hv) {
  // 16-bit A-matrix 16x32: slots 0-3 -> K=2s (+8*half), slots 4-7 -> K=16+2(s-4) (+8*half)
  return (s < 4 ? 2 * s : 16 + 2 * (s - 4)) + 8 * hv;
}

// A-fragment (16x32 bf16), rows contiguous with stride ld. Also valid as a
// B-fragment when the 32x16 B matrix is stored with row index = N column.
__device__ __forceinline__ v16bf load_afrag(const __bf16* __restrict__ p0, int ld, int r0, int k0) {
  const int lane = threadIdx.x & 31;
  const int m = lane & 15, hv = lane >> 4;
  const __bf16* p = p0 + (size_t)(r0 + m) * ld + k0 + 8 * hv;
  v16bf a;
#pragma unroll
  for (int i = 0; i < 8; ++i) a[i] = p[i];          // K = 8*half .. +7
#pragma unroll
  for (int i = 0; i < 8; ++i) a[8 + i] = p[16 + i]; // K = 16+8*half .. +7
  return a;
}

// B-fragment (32x16 bf16) from a row-major [K][N] LDS tile (stride ld): lane = column n.
__device__ __forceinline__ v16bf load_bfrag_col(const __bf16* __restrict__ p0, int ld, int k0, int n0) {
  const int lane = threadIdx.x & 31;
  const int n = lane & 15, hv = lane >> 4;
  v16bf b;
#pragma unroll
  for (int s = 0; s < 8; ++s) {
    const int k = kpack_base(s, hv);
    b[2 * s]     = p0[(size_t)(k0 + k) * ld + n0 + n];
    b[2 * s + 1] = p0[(size_t)(k0 + k + 1) * ld + n0 + n];
  }
  return b;
}

__device__ __forceinline__ v8f wmma_bf16(v16bf a, v16bf b, v8f c) {
  return __builtin_amdgcn_wmma_f32_16x16x32_bf16(false, a, false, b, (short)0, c, false, false);
}

// ---------------- elementwise kernels ----------------

__global__ void f2bf_kernel(const float* __restrict__ a, __bf16* __restrict__ o, int n) {
  int i = blockIdx.x * blockDim.x + threadIdx.x;
  const int stride = gridDim.x * blockDim.x;
  for (; i < n; i += stride) o[i] = (__bf16)a[i];
}

__global__ __launch_bounds__(256) void layernorm_bf16_kernel(
    const float* __restrict__ x, const float* __restrict__ w,
    const float* __restrict__ b, __bf16* __restrict__ out) {
  const int row = blockIdx.x, tid = threadIdx.x;
  const int lane = tid & 31, wid = tid >> 5;
  const float4 v = reinterpret_cast<const float4*>(x + (size_t)row * kD)[tid];
  float s  = v.x + v.y + v.z + v.w;
  float s2 = v.x * v.x + v.y * v.y + v.z * v.z + v.w * v.w;
#pragma unroll
  for (int m = 1; m < 32; m <<= 1) { s += __shfl_xor(s, m); s2 += __shfl_xor(s2, m); }
  __shared__ float red[16];
  if (lane == 0) { red[wid] = s; red[8 + wid] = s2; }
  __syncthreads();
  float ts = 0.f, ts2 = 0.f;
#pragma unroll
  for (int i = 0; i < 8; ++i) { ts += red[i]; ts2 += red[8 + i]; }
  const float mu  = ts * (1.0f / kD);
  const float var = ts2 * (1.0f / kD) - mu * mu;
  const float r   = rsqrtf(var + 1e-5f);
  const float4 wv = reinterpret_cast<const float4*>(w)[tid];
  const float4 bv = reinterpret_cast<const float4*>(b)[tid];
  __bf16* po = out + (size_t)row * kD + tid * 4;
  po[0] = (__bf16)((v.x - mu) * r * wv.x + bv.x);
  po[1] = (__bf16)((v.y - mu) * r * wv.y + bv.y);
  po[2] = (__bf16)((v.z - mu) * r * wv.z + bv.z);
  po[3] = (__bf16)((v.w - mu) * r * wv.w + bv.w);
}

// ---------------- tiled WMMA GEMM, async-staged + double-buffered ----------------

enum { MODE_Q = 0, MODE_KVX = 1, MODE_KVL = 2, MODE_OUT = 3 };

template <int MODE>
__global__ __launch_bounds__(256) void gemm_bf16_kernel(
    const __bf16* __restrict__ A, const __bf16* __restrict__ Bw,
    int Kdim, int Ncols, int rows_per_batch, int kv_row_off,
    __bf16* __restrict__ outK, __bf16* __restrict__ outV,
    __bf16* __restrict__ outQ, float* __restrict__ outF) {
  __shared__ __bf16 lsA[2][128 * 40];  // A tile 128x32, row stride 40 (bank padding)
  __shared__ __bf16 lsB[2][32 * 136];  // B tile 32x128 natural [k][n] order, stride 136
  const int tid = threadIdx.x;
  const int lane = tid & 31, wid = tid >> 5;
  const int row0 = blockIdx.x * 128, col0 = blockIdx.y * 128;
  const int wr = (wid >> 2) * 64, wc = (wid & 3) * 32;

  v8f acc[4][2];
#pragma unroll
  for (int a = 0; a < 4; ++a)
#pragma unroll
    for (int b = 0; b < 2; ++b) acc[a][b] = (v8f){0.f, 0.f, 0.f, 0.f, 0.f, 0.f, 0.f, 0.f};

  auto stage = [&](int buf, int k0) {
#pragma unroll
    for (int i = 0; i < 2; ++i) {  // A: 128x32 = 4096 elems, 16B per lane per iter
      const int lin = i * 2048 + tid * 8;
      const int r = lin >> 5, c = lin & 31;
      async_load_b128(A + (size_t)(row0 + r) * Kdim + k0 + c,
                      lds_off(&lsA[buf][r * 40 + c]));
    }
#pragma unroll
    for (int i = 0; i < 2; ++i) {  // B: 32x128 = 4096 elems
      const int lin = i * 2048 + tid * 8;
      const int kk = lin >> 7, c = lin & 127;
      async_load_b128(Bw + (size_t)(k0 + kk) * Ncols + col0 + c,
                      lds_off(&lsB[buf][kk * 136 + c]));
    }
  };

  stage(0, 0);
  wait_async0();
  __syncthreads();

  const int nsteps = Kdim >> 5;
  for (int kt = 0; kt < nsteps; ++kt) {
    const int buf = kt & 1;
    if (kt + 1 < nsteps) stage(buf ^ 1, (kt + 1) << 5);  // prefetch next tile
    v16bf bf[2];
#pragma unroll
    for (int ni = 0; ni < 2; ++ni) bf[ni] = load_bfrag_col(&lsB[buf][0], 136, 0, wc + 16 * ni);
#pragma unroll
    for (int mi = 0; mi < 4; ++mi) {
      const v16bf af = load_afrag(&lsA[buf][0], 40, wr + 16 * mi, 0);
#pragma unroll
      for (int ni = 0; ni < 2; ++ni) acc[mi][ni] = wmma_bf16(af, bf[ni], acc[mi][ni]);
    }
    wait_async0();
    __syncthreads();
  }

  // ---- epilogue: D-frag layout: value(vgpr i, lane) = (M = i + 8*half, N = lane&15)
  const int nl = lane & 15, hv = lane >> 4;

  if constexpr (MODE == MODE_Q) {
#pragma unroll
    for (int mi = 0; mi < 4; ++mi)
#pragma unroll
      for (int ni = 0; ni < 2; ++ni)
#pragma unroll
        for (int i = 0; i < 8; ++i) {
          const int grow = row0 + wr + mi * 16 + i + 8 * hv;
          const int gcol = col0 + wc + ni * 16 + nl;
          const int b_ = grow >> 8, m_ = grow & 255;  // kM = 256
          const int h_ = gcol >> 6, hd = gcol & 63;
          outQ[(((size_t)b_ * kH + h_) * kM + m_) * kHD + hd] =
              (__bf16)(acc[mi][ni][i] * 0.125f);  // HD^-0.5
        }
  } else if constexpr (MODE == MODE_OUT) {
#pragma unroll
    for (int mi = 0; mi < 4; ++mi)
#pragma unroll
      for (int ni = 0; ni < 2; ++ni)
#pragma unroll
        for (int i = 0; i < 8; ++i) {
          const int grow = row0 + wr + mi * 16 + i + 8 * hv;
          const int gcol = col0 + wc + ni * 16 + nl;
          outF[(size_t)grow * Ncols + gcol] = acc[mi][ni][i];
        }
  } else {
    const int cb = col0 + wc;
    const bool rope = (MODE == MODE_KVX) && (cb < kInner) && ((cb & 63) == 0);
    if (rope) {
      // This wave covers hd 0..31 of one head: frag ni=0 holds d=0..15, ni=1 holds d+16.
      const int h_ = cb >> 6;
      const int d = nl;                 // 0..15
      const int j = d & 7;              // freq index
      const float invj = __expf(-1.1512925465f * (float)j);  // 10000^(-j/8)
#pragma unroll
      for (int mi = 0; mi < 4; ++mi)
#pragma unroll
        for (int i = 0; i < 8; ++i) {
          const int grow = row0 + wr + mi * 16 + i + 8 * hv;
          const int b_ = grow / rows_per_batch, n_ = grow % rows_per_batch;
          const float pos = (d < 8) ? (float)(n_ >> 6) : (float)(n_ & 63);  // 64x64 grid
          const float f = pos * invj;
          const float cs = __cosf(f), sn = __sinf(f);
          const float v0 = acc[mi][0][i], v1 = acc[mi][1][i];
          const size_t base = (((size_t)b_ * kH + h_) * kNM + kv_row_off + n_) * kHD;
          outK[base + d]      = (__bf16)(v0 * cs - v1 * sn);  // rotate_half: [-x2, x1]
          outK[base + d + 16] = (__bf16)(v1 * cs + v0 * sn);
        }
    } else {
#pragma unroll
      for (int mi = 0; mi < 4; ++mi)
#pragma unroll
        for (int ni = 0; ni < 2; ++ni)
#pragma unroll
          for (int i = 0; i < 8; ++i) {
            const int grow = row0 + wr + mi * 16 + i + 8 * hv;
            const int gcol = col0 + wc + ni * 16 + nl;
            const int b_ = grow / rows_per_batch, n_ = grow % rows_per_batch;
            __bf16* dst; int c_;
            if (gcol < kInner) { dst = outK; c_ = gcol; }
            else               { dst = outV; c_ = gcol - kInner; }
            const int h_ = c_ >> 6, hd = c_ & 63;
            dst[(((size_t)b_ * kH + h_) * kNM + kv_row_off + n_) * kHD + hd] =
                (__bf16)acc[mi][ni][i];
          }
    }
  }
}

// ---------------- flash attention (per (b,h), 128 queries / block) ----------------

__global__ __launch_bounds__(256) void attn_kernel(
    const __bf16* __restrict__ Q, const __bf16* __restrict__ Kg,
    const __bf16* __restrict__ Vg, __bf16* __restrict__ O) {
  __shared__ __bf16 lsK[2][128 * 72];
  __shared__ __bf16 lsV[2][128 * 72];
  __shared__ __bf16 lsP[8][16 * 136];  // per-wave D-frag -> A-frag relayout patch
  const int tid = threadIdx.x, lane = tid & 31, wid = tid >> 5;
  const int nl = lane & 15, hv = lane >> 4;
  const int bh = blockIdx.x >> 1;
  const int qr = (blockIdx.x & 1) * 128 + wid * 16;  // this wave's 16 query rows

  // Q resident in A-fragments (K = hd, two chunks of 32)
  const __bf16* qb = Q + ((size_t)bh * kM + qr) * kHD;
  v16bf qa[2];
#pragma unroll
  for (int c = 0; c < 2; ++c) {
    const __bf16* p = qb + (size_t)nl * kHD + c * 32 + 8 * hv;
#pragma unroll
    for (int i = 0; i < 8; ++i) qa[c][i] = p[i];
#pragma unroll
    for (int i = 0; i < 8; ++i) qa[c][8 + i] = p[16 + i];
  }

  float mrow[8], lrow[8];
  v8f o[4];
#pragma unroll
  for (int i = 0; i < 8; ++i) { mrow[i] = -1e30f; lrow[i] = 0.f; }
#pragma unroll
  for (int t = 0; t < 4; ++t) o[t] = (v8f){0.f, 0.f, 0.f, 0.f, 0.f, 0.f, 0.f, 0.f};

  const __bf16* kgb = Kg + (size_t)bh * kNM * kHD;
  const __bf16* vgb = Vg + (size_t)bh * kNM * kHD;

  auto stage_kv = [&](int buf, int tk) {
    const size_t kb = (size_t)tk * 128;
#pragma unroll
    for (int i = 0; i < 4; ++i) {  // 128x64 per tile, 16B per lane per iter
      const int lin = i * 2048 + tid * 8;
      const int r = lin >> 6, c = lin & 63;
      async_load_b128(kgb + (kb + r) * kHD + c, lds_off(&lsK[buf][r * 72 + c]));
      async_load_b128(vgb + (kb + r) * kHD + c, lds_off(&lsV[buf][r * 72 + c]));
    }
  };

  constexpr int kTiles = kNM / 128;  // 34
  stage_kv(0, 0);
  wait_async0();
  __syncthreads();

  for (int tk = 0; tk < kTiles; ++tk) {
    const int buf = tk & 1;
    if (tk + 1 < kTiles) stage_kv(buf ^ 1, tk + 1);  // prefetch next K/V tile

    // S = Q (16x64) @ K^T (64x128): K rows are keys -> B-frag == A-frag pattern
    v8f s[8];
#pragma unroll
    for (int j2 = 0; j2 < 8; ++j2) {
      s[j2] = (v8f){0.f, 0.f, 0.f, 0.f, 0.f, 0.f, 0.f, 0.f};
#pragma unroll
      for (int c = 0; c < 2; ++c) {
        const v16bf kf = load_afrag(&lsK[buf][0], 72, j2 * 16, c * 32);
        s[j2] = wmma_bf16(qa[c], kf, s[j2]);
      }
    }

    // online softmax per query row (row = i + 8*half; cols spread over 16 lanes)
    float corr[8];
#pragma unroll
    for (int i = 0; i < 8; ++i) {
      float tm = s[0][i];
#pragma unroll
      for (int j2 = 1; j2 < 8; ++j2) tm = fmaxf(tm, s[j2][i]);
#pragma unroll
      for (int msk = 1; msk < 16; msk <<= 1) tm = fmaxf(tm, __shfl_xor(tm, msk));
      const float mn = fmaxf(mrow[i], tm);
      corr[i] = __expf(mrow[i] - mn);
      mrow[i] = mn;
      float rs = 0.f;
#pragma unroll
      for (int j2 = 0; j2 < 8; ++j2) {
        const float pv = __expf(s[j2][i] - mn);
        s[j2][i] = pv;
        rs += pv;
      }
#pragma unroll
      for (int msk = 1; msk < 16; msk <<= 1) rs += __shfl_xor(rs, msk);
      lrow[i] = lrow[i] * corr[i] + rs;
    }
#pragma unroll
    for (int t = 0; t < 4; ++t)
#pragma unroll
      for (int i = 0; i < 8; ++i) o[t][i] *= corr[i];

    // stage P (D-frag layout) into per-wave LDS, reload as A-frags
    __bf16* pw = &lsP[wid][0];
#pragma unroll
    for (int j2 = 0; j2 < 8; ++j2)
#pragma unroll
      for (int i = 0; i < 8; ++i)
        pw[(i + 8 * hv) * 136 + j2 * 16 + nl] = (__bf16)s[j2][i];

    // O += P (16x128) @ V (128x64)
#pragma unroll
    for (int c = 0; c < 4; ++c) {
      const v16bf pa = load_afrag(pw, 136, 0, c * 32);
#pragma unroll
      for (int t = 0; t < 4; ++t) {
        const v16bf bv = load_bfrag_col(&lsV[buf][0], 72, c * 32, t * 16);
        o[t] = wmma_bf16(pa, bv, o[t]);
      }
    }
    wait_async0();
    __syncthreads();
  }

  // finalize: divide by l, write O as (B, M, H*HD) bf16 for the Wo GEMM
  const int b_ = bh >> 4, h_ = bh & 15;
#pragma unroll
  for (int i = 0; i < 8; ++i) {
    const float inv = 1.0f / lrow[i];
    const int m_ = qr + i + 8 * hv;
#pragma unroll
    for (int t = 0; t < 4; ++t)
      O[((size_t)b_ * kM + m_) * kInner + h_ * kHD + t * 16 + nl] =
          (__bf16)(o[t][i] * inv);
  }
}

// ---------------- launch ----------------

extern "C" void kernel_launch(void* const* d_in, const int* in_sizes, int n_in,
                              void* d_out, int out_size, void* d_ws, size_t ws_size,
                              hipStream_t stream) {
  const float* x   = (const float*)d_in[0];
  const float* lat = (const float*)d_in[1];
  const float* lxw = (const float*)d_in[2];
  const float* lxb = (const float*)d_in[3];
  const float* llw = (const float*)d_in[4];
  const float* llb = (const float*)d_in[5];
  const float* Wq  = (const float*)d_in[6];
  const float* Wkv = (const float*)d_in[7];
  const float* Wo  = (const float*)d_in[8];
  float* out = (float*)d_out;

  char* p = (char*)d_ws;
  auto take = [&p](size_t elems) {
    __bf16* r = (__bf16*)p;
    p += ((elems * sizeof(__bf16)) + 255) & ~(size_t)255;
    return r;
  };
  __bf16* xn   = take((size_t)kB * kN * kD);
  __bf16* lnl  = take((size_t)kB * kM * kD);
  __bf16* wqb  = take((size_t)kD * kInner);
  __bf16* wkvb = take((size_t)kD * 2 * kInner);
  __bf16* wob  = take((size_t)kInner * kD);
  __bf16* qbuf = take((size_t)kB * kH * kM * kHD);
  __bf16* kbuf = take((size_t)kB * kH * kNM * kHD);
  __bf16* vbuf = take((size_t)kB * kH * kNM * kHD);
  __bf16* obuf = take((size_t)kB * kM * kInner);

  f2bf_kernel<<<1024, 256, 0, stream>>>(Wq, wqb, kD * kInner);
  f2bf_kernel<<<2048, 256, 0, stream>>>(Wkv, wkvb, kD * 2 * kInner);
  f2bf_kernel<<<1024, 256, 0, stream>>>(Wo, wob, kInner * kD);
  layernorm_bf16_kernel<<<kB * kN, 256, 0, stream>>>(x, lxw, lxb, xn);
  layernorm_bf16_kernel<<<kB * kM, 256, 0, stream>>>(lat, llw, llb, lnl);

  gemm_bf16_kernel<MODE_Q><<<dim3((kB * kM) / 128, kInner / 128), 256, 0, stream>>>(
      lnl, wqb, kD, kInner, kM, 0, nullptr, nullptr, qbuf, nullptr);
  gemm_bf16_kernel<MODE_KVX><<<dim3((kB * kN) / 128, (2 * kInner) / 128), 256, 0, stream>>>(
      xn, wkvb, kD, 2 * kInner, kN, 0, kbuf, vbuf, nullptr, nullptr);
  gemm_bf16_kernel<MODE_KVL><<<dim3((kB * kM) / 128, (2 * kInner) / 128), 256, 0, stream>>>(
      lnl, wkvb, kD, 2 * kInner, kM, kN, kbuf, vbuf, nullptr, nullptr);
  attn_kernel<<<kB * kH * (kM / 128), 256, 0, stream>>>(qbuf, kbuf, vbuf, obuf);
  gemm_bf16_kernel<MODE_OUT><<<dim3((kB * kM) / 128, kD / 128), 256, 0, stream>>>(
      obuf, wob, kInner, kD, kM, 0, nullptr, nullptr, nullptr, out);
}
